// ConvVQVAE_12902081757262
// MI455X (gfx1250) — compile-verified, matrix operand order checked by reading, not run on hardware
//
#include <hip/hip_runtime.h>
#include <hip/hip_bf16.h>
#include <math.h>

typedef __attribute__((ext_vector_type(16))) _Float16 v16h;
typedef __attribute__((ext_vector_type(8)))  _Float16 h8;
typedef __attribute__((ext_vector_type(8)))  float    v8f;

__device__ __forceinline__ float gelu_exact(float v) {
    return 0.5f * v * (1.f + erff(v * 0.70710678118654752f));
}

// ---------------------------------------------------------------------------
// Weight fragment prep: reorganize OIHW (or deconv IOHW+flip) fp32 weights into
// WMMA B-fragment layout, fp16.  Global layout: [ntile][tap][kc][lane*16+e]
// B 32x16 (KxN) lane map (mirror of A): lane<16 -> N=lane, K elems {b..b+7, 16+b..16+b+7}
// with b = (lane<16 ? 0 : 8).
// ---------------------------------------------------------------------------
__global__ void __launch_bounds__(256) vqvae_prep_wfrag(
    const float* __restrict__ w, _Float16* __restrict__ bf,
    int O, int I, int ksize, int deconv)
{
    const int taps = ksize * ksize;
    const int kc = I >> 5;
    const int ntiles = O >> 4;
    const int total = ntiles * taps * kc * 512;
    for (int i = blockIdx.x * 256 + threadIdx.x; i < total; i += gridDim.x * 256) {
        int e = i & 15;
        int l = (i >> 4) & 31;
        int rest = i >> 9;
        int c = rest % kc; rest /= kc;
        int t = rest % taps; rest /= taps;
        int nt = rest;
        int kbase = (l < 16) ? 0 : 8;
        int kk = (e < 8) ? (kbase + e) : (16 + kbase + (e - 8));
        int ci = c * 32 + kk;
        int o = nt * 16 + (l & 15);
        int ky = t / ksize, kx = t - ky * ksize;
        float val;
        if (!deconv)
            val = w[(((size_t)o * I + ci) * ksize + ky) * ksize + kx];
        else  // torch ConvTranspose2d layout (in,out,kh,kw), spatially flipped
            val = w[(((size_t)ci * O + o) * ksize + (ksize - 1 - ky)) * ksize + (ksize - 1 - kx)];
        bf[((((size_t)nt * taps + t) * kc + c) << 9) + (l << 4) + e] = (_Float16)val;
    }
}

__global__ void vqvae_prep_e2(const float* __restrict__ E, float* __restrict__ e2, int K, int D)
{
    int n = blockIdx.x * blockDim.x + threadIdx.x;
    if (n >= K) return;
    float s = 0.f;
    for (int d = 0; d < D; ++d) { float v = E[(size_t)n * D + d]; s += v * v; }
    e2[n] = s;
}

__global__ void vqvae_zero_acc(float* acc) { if (threadIdx.x < 2) acc[threadIdx.x] = 0.f; }

// ---------------------------------------------------------------------------
// Generic implicit-GEMM conv / dilated-input deconv, NHWC fp16, WMMA f16->f32.
// Fully unrolled over taps and K-chunks via template params -> straight-line
// runs of v_wmma_f32_16x16x32_f16 per tile.
// One wave32 = one 16-pixel x 16-cout tile.  blockIdx.y = cout tile.
// act: 0=none, 1=relu, 2=gelu, 3=relu then gelu (basic-block tail)
// ---------------------------------------------------------------------------
template<int KSIZE, int KC, int DECONV>
__global__ void __launch_bounds__(256) vqvae_conv_wmma(
    const _Float16* __restrict__ in,     // [Bn,Hin,Win,Cin] fp16
    const _Float16* __restrict__ bfrag,  // prepped fragments
    const float* __restrict__ bn_g, const float* __restrict__ bn_b,
    const _Float16* __restrict__ skip,   // [Bn,Hout,Wout,Cout] fp16 or null
    _Float16* __restrict__ out_h, float* __restrict__ out_f,
    int Bn, int Hin, int Win, int Hout, int Wout, int Cout,
    int stride, int pad, int act)
{
    constexpr int TAPS = KSIZE * KSIZE;
    constexpr int CIN  = KC * 32;
    constexpr int NELEMS = TAPS * KC * 512;
    __shared__ __align__(32) _Float16 lds_b[NELEMS];
    const int lane = threadIdx.x & 31;
    const int wave = threadIdx.x >> 5;

    const int M = Bn * Hout * Wout;
    const int tile = blockIdx.x * 8 + wave;
    const int m = lane & 15;
    const int p = tile * 16 + m;
    const bool prow = (p < M);
    int b = 0, oy = 0, ox = 0;
    if (prow) {
        b = p / (Hout * Wout);
        int r = p - b * Hout * Wout;
        oy = r / Wout; ox = r - oy * Wout;
    }
    // prefetch the (approximate) center-tap activation row while LDS fills
    if (prow) {
        int cy = DECONV ? (oy >> 1) : (oy * stride);
        int cx = DECONV ? (ox >> 1) : (ox * stride);
        cy = cy < 0 ? 0 : (cy >= Hin ? Hin - 1 : cy);
        cx = cx < 0 ? 0 : (cx >= Win ? Win - 1 : cx);
        __builtin_prefetch(in + (((size_t)(b * Hin + cy) * Win + cx) * CIN), 0, 0);
    }

    // cooperative 128-bit staging of this cout-tile's weight fragments
    {
        const h8* gsrc = (const h8*)(bfrag + (size_t)blockIdx.y * NELEMS);
        h8* ldst = (h8*)lds_b;
#pragma unroll
        for (int i = threadIdx.x; i < NELEMS / 8; i += 256) ldst[i] = gsrc[i];
    }
    __syncthreads();

    if (tile * 16 >= M) return;

    const int sel = (lane < 16) ? 0 : 1;   // kbase/8
    const h8 hzero = {};

    v8f acc = {};
#pragma unroll
    for (int t = 0; t < TAPS; ++t) {
        const int ky = t / KSIZE, kx = t - ky * KSIZE;
        int sy, sx; bool valid = prow;
        if (!DECONV) {
            sy = oy * stride - pad + ky;
            sx = ox * stride - pad + kx;
            valid = valid && sy >= 0 && sy < Hin && sx >= 0 && sx < Win;
        } else {  // lhs_dilation=2, pad = pad_lo
            int iy = oy + ky - pad, ix = ox + kx - pad;
            valid = valid && iy >= 0 && ix >= 0 && !(iy & 1) && !(ix & 1);
            sy = iy >> 1; sx = ix >> 1;
            valid = valid && sy < Hin && sx < Win;
        }
        const _Float16* src = in + (((size_t)(b * Hin + sy) * Win + sx) * CIN);
#pragma unroll
        for (int c = 0; c < KC; ++c) {
            union { v16h v; h8 h2[2]; } a;
            if (valid) {
                const h8* sp = (const h8*)(src + c * 32);
                a.h2[0] = sp[sel];       // K kbase..kbase+7
                a.h2[1] = sp[2 + sel];   // K 16+kbase..16+kbase+7
            } else { a.h2[0] = hzero; a.h2[1] = hzero; }
            const v16h bf = *(const v16h*)&lds_b[((t * KC + c) << 9) + (lane << 4)];
            acc = __builtin_amdgcn_wmma_f32_16x16x32_f16(false, a.v, false, bf,
                                                         (short)0, acc, false, false);
        }
    }

    // epilogue: C/D layout -> lane<16: N=lane, M=r ; lane>=16: N=lane-16, M=8+r
    const int n = blockIdx.y * 16 + (lane & 15);
    float scale = 1.f, bias = 0.f;
    if (bn_g) { scale = bn_g[n] * rsqrtf(1.f + 1e-5f); bias = bn_b[n]; }
    const int mbase = (lane < 16) ? 0 : 8;
#pragma unroll
    for (int r = 0; r < 8; ++r) {
        int pp = tile * 16 + mbase + r;
        if (pp < M) {
            float v = acc[r];
            if (bn_g) v = v * scale + bias;
            if (skip) v += (float)skip[(size_t)pp * Cout + n];
            if (act == 1 || act == 3) v = fmaxf(v, 0.f);
            if (act == 2 || act == 3) v = gelu_exact(v);
            if (out_h) out_h[(size_t)pp * Cout + n] = (_Float16)v;
            if (out_f) out_f[(size_t)pp * Cout + n] = v;
        }
    }
}

// ---------------------------------------------------------------------------
// Encoder conv1 (Cin=3) direct VALU: NCHW fp32 in -> NHWC fp16 + GELU
// ---------------------------------------------------------------------------
__global__ void __launch_bounds__(256) vqvae_conv_ec1(
    const float* __restrict__ x, const float* __restrict__ w,
    _Float16* __restrict__ out, int Bn)
{
    __shared__ float ws[32 * 27];
    for (int i = threadIdx.x; i < 32 * 27; i += 256) ws[i] = w[i];
    __syncthreads();
    const int total = Bn * 64 * 64 * 32;
    for (int i = blockIdx.x * 256 + threadIdx.x; i < total; i += gridDim.x * 256) {
        int co = i & 31;
        int p = i >> 5;
        int b = p >> 12;              // /4096
        int r = p & 4095;
        int oy = r >> 6, ox = r & 63;
        float s = 0.f;
        for (int ci = 0; ci < 3; ++ci)
#pragma unroll
            for (int ky = 0; ky < 3; ++ky) {
                int iy = oy * 2 - 1 + ky;
                if (iy < 0 || iy >= 128) continue;
#pragma unroll
                for (int kx = 0; kx < 3; ++kx) {
                    int ix = ox * 2 - 1 + kx;
                    if (ix < 0 || ix >= 128) continue;
                    s += x[(((size_t)b * 3 + ci) * 128 + iy) * 128 + ix] *
                         ws[((co * 3 + ci) * 3 + ky) * 3 + kx];
                }
            }
        out[i] = (_Float16)gelu_exact(s);
    }
}

// ---------------------------------------------------------------------------
// VQ: WMMA GEMM z(16px x 64) . E^T(64 x 512) with fused argmin, z_q gather,
// one-hot emit and codebook-loss partial sums.  E fragments staged in LDS in
// two 32KB phases, shared by all 8 waves of the block (8x less L2 traffic).
// ---------------------------------------------------------------------------
__global__ void __launch_bounds__(256) vqvae_vq(
    const _Float16* __restrict__ ze_h, const float* __restrict__ ze_f,
    const float* __restrict__ E, const _Float16* __restrict__ efrag,
    const float* __restrict__ e2,
    _Float16* __restrict__ zq_h, float* __restrict__ out_oh,
    float* __restrict__ out_idx, int* __restrict__ idx_i,
    float* __restrict__ acc_vq, int M)
{
    __shared__ __align__(32) _Float16 lds_e[16 * 1024];  // 32 KB: 16 code-tiles/phase
    __shared__ float lds_e2[512];
    const int lane = threadIdx.x & 31;
    const int wave = threadIdx.x >> 5;
    const int tile = blockIdx.x * 8 + wave;
    const bool active = (tile * 16) < M;
    const int sel = (lane < 16) ? 0 : 1;
    const int m = lane & 15;
    int p = tile * 16 + m;
    if (p >= M) p = M - 1;           // clamp loads; stores guarded by `active`

    for (int i = threadIdx.x; i < 512; i += 256) lds_e2[i] = e2[i];

    const h8* zp = (const h8*)(ze_h + (size_t)p * 64);
    union { v16h v; h8 h2[2]; } a0, a1;
    a0.h2[0] = zp[sel];     a0.h2[1] = zp[2 + sel];   // K 0..31
    a1.h2[0] = zp[4 + sel]; a1.h2[1] = zp[6 + sel];   // K 32..63

    float rmin[8]; int ridx[8];
#pragma unroll
    for (int r = 0; r < 8; ++r) { rmin[r] = 3.4e38f; ridx[r] = 0; }

    for (int phase = 0; phase < 2; ++phase) {
        __syncthreads();             // prior phase fully consumed
        {
            const h8* gsrc = (const h8*)(efrag + (size_t)phase * 16 * 1024);
            h8* ldst = (h8*)lds_e;
#pragma unroll
            for (int i = threadIdx.x; i < 2048; i += 256) ldst[i] = gsrc[i];
        }
        __syncthreads();
#pragma unroll 2
        for (int ntl = 0; ntl < 16; ++ntl) {
            const v16h b0 = *(const v16h*)&lds_e[(ntl * 2 + 0) * 512 + (lane << 4)];
            const v16h b1 = *(const v16h*)&lds_e[(ntl * 2 + 1) * 512 + (lane << 4)];
            v8f acc = {};
            acc = __builtin_amdgcn_wmma_f32_16x16x32_f16(false, a0.v, false, b0, (short)0, acc, false, false);
            acc = __builtin_amdgcn_wmma_f32_16x16x32_f16(false, a1.v, false, b1, (short)0, acc, false, false);
            const int n = (phase * 16 + ntl) * 16 + (lane & 15);
            const float en = lds_e2[n];
#pragma unroll
            for (int r = 0; r < 8; ++r) {
                float d = en - 2.f * acc[r];   // ||z||^2 row-constant: argmin-invariant
                if (d < rmin[r]) { rmin[r] = d; ridx[r] = n; }
            }
        }
    }
    if (!active) return;

    // reduce across the 16 lanes of each row group (lanes 0-15: rows 0-7, 16-31: rows 8-15)
#pragma unroll
    for (int r = 0; r < 8; ++r)
        for (int mask = 8; mask >= 1; mask >>= 1) {
            float om = __shfl_xor(rmin[r], mask, 32);
            int   oi = __shfl_xor(ridx[r], mask, 32);
            if (om < rmin[r] || (om == rmin[r] && oi < ridx[r])) { rmin[r] = om; ridx[r] = oi; }
        }

    float lsum = 0.f;
    for (int mm = 0; mm < 16; ++mm) {
        const int r = mm & 7;
        const int srcl = (mm < 8) ? 0 : 16;
        const int idx = __shfl(ridx[r], srcl, 32);
        const int pm = tile * 16 + mm;
        if (pm < M) {
            if (lane == srcl) { out_idx[pm] = (float)idx; idx_i[pm] = idx; }
            const float* ev = E + (size_t)idx * 64;
#pragma unroll
            for (int j = 0; j < 2; ++j) {
                int ch = lane * 2 + j;
                float evv = ev[ch];
                zq_h[(size_t)pm * 64 + ch] = (_Float16)evv;
                float diff = ze_f[(size_t)pm * 64 + ch] - evv;
                lsum += diff * diff;
            }
            float* oh = out_oh + (size_t)pm * 512 + lane * 16;
#pragma unroll
            for (int j = 0; j < 16; ++j) oh[j] = (lane * 16 + j == idx) ? 1.f : 0.f;
        }
    }
    for (int mask = 16; mask >= 1; mask >>= 1) lsum += __shfl_xor(lsum, mask, 32);
    if (lane == 0) atomicAdd(acc_vq, lsum);
}

// ---------------------------------------------------------------------------
// Final deconv (32->3, k3 s2 pads(0,1)) + sigmoid + x_pred(NCHW) + MSE partials
// ---------------------------------------------------------------------------
__global__ void __launch_bounds__(256) vqvae_deconv_final(
    const _Float16* __restrict__ hin,  // [Bn,65,65,32] fp16
    const float* __restrict__ w,       // dt2 raw (32,3,3,3)
    const float* __restrict__ x,       // input NCHW
    float* __restrict__ xpred, float* __restrict__ acc1, int Bn)
{
    __shared__ float ws[9][32][3];     // wt[ky*3+kx][ci][o]
    for (int i = threadIdx.x; i < 9 * 32 * 3; i += 256) {
        int o = i % 3; int rest = i / 3; int ci = rest % 32; int t = rest / 32;
        int ky = t / 3, kx = t % 3;
        ws[t][ci][o] = w[(((size_t)ci * 3 + o) * 3 + (2 - ky)) * 3 + (2 - kx)];
    }
    __syncthreads();
    const int total = Bn * 128 * 128;
    float lsum = 0.f;
    for (int i = blockIdx.x * 256 + threadIdx.x; i < total; i += gridDim.x * 256) {
        int b = i >> 14;
        int r = i & 16383;
        int oy = r >> 7, ox = r & 127;
        float s0 = 0.f, s1 = 0.f, s2 = 0.f;
#pragma unroll
        for (int ky = 0; ky < 3; ++ky) {
            int iy = oy + ky;                  // pad_lo = 0
            if (iy & 1) continue;
            int sy = iy >> 1; if (sy >= 65) continue;
#pragma unroll
            for (int kx = 0; kx < 3; ++kx) {
                int ix = ox + kx;
                if (ix & 1) continue;
                int sx = ix >> 1; if (sx >= 65) continue;
                const _Float16* a = hin + (((size_t)b * 65 + sy) * 65 + sx) * 32;
                const float* wv = &ws[ky * 3 + kx][0][0];
#pragma unroll
                for (int ci = 0; ci < 32; ++ci) {
                    float av = (float)a[ci];
                    s0 += av * wv[ci * 3 + 0];
                    s1 += av * wv[ci * 3 + 1];
                    s2 += av * wv[ci * 3 + 2];
                }
            }
        }
        float p0 = 1.f / (1.f + expf(-s0));
        float p1 = 1.f / (1.f + expf(-s1));
        float p2 = 1.f / (1.f + expf(-s2));
        size_t o0 = (((size_t)b * 3 + 0) * 128 + oy) * 128 + ox;
        xpred[o0] = p0; xpred[o0 + 16384] = p1; xpred[o0 + 32768] = p2;
        float d0 = x[o0] - p0, d1 = x[o0 + 16384] - p1, d2 = x[o0 + 32768] - p2;
        lsum += d0 * d0 + d1 * d1 + d2 * d2;
    }
    for (int mask = 16; mask >= 1; mask >>= 1) lsum += __shfl_xor(lsum, mask, 32);
    if ((threadIdx.x & 31) == 0) atomicAdd(acc1, lsum);
}

__global__ void vqvae_finalize(const float* acc, float* loss_out, float n1, float n2, float invB)
{
    if (threadIdx.x == 0)
        loss_out[0] = (acc[0] / n1 + 1.25f * acc[1] / n2) * invB;
}

// ===========================================================================
extern "C" void kernel_launch(void* const* d_in, const int* in_sizes, int n_in,
                              void* d_out, int out_size, void* d_ws, size_t ws_size,
                              hipStream_t stream)
{
    const float* x     = (const float*)d_in[0];
    const float* Emb   = (const float*)d_in[1];
    const float* w_ec1 = (const float*)d_in[2];
    const float* w_ec2 = (const float*)d_in[3];
    const float* be1c1 = (const float*)d_in[4];
    const float* be1g1 = (const float*)d_in[5];  const float* be1b1 = (const float*)d_in[6];
    const float* be1c2 = (const float*)d_in[7];
    const float* be1g2 = (const float*)d_in[8];  const float* be1b2 = (const float*)d_in[9];
    const float* be2c1 = (const float*)d_in[10];
    const float* be2g1 = (const float*)d_in[11]; const float* be2b1 = (const float*)d_in[12];
    const float* be2c2 = (const float*)d_in[13];
    const float* be2g2 = (const float*)d_in[14]; const float* be2b2 = (const float*)d_in[15];
    const float* w_ec3 = (const float*)d_in[16];
    const float* bd1c1 = (const float*)d_in[17];
    const float* bd1g1 = (const float*)d_in[18]; const float* bd1b1 = (const float*)d_in[19];
    const float* bd1c2 = (const float*)d_in[20];
    const float* bd1g2 = (const float*)d_in[21]; const float* bd1b2 = (const float*)d_in[22];
    const float* bd1sc = (const float*)d_in[23];
    const float* bd1sg = (const float*)d_in[24]; const float* bd1sb = (const float*)d_in[25];
    const float* bd2c1 = (const float*)d_in[26];
    const float* bd2g1 = (const float*)d_in[27]; const float* bd2b1 = (const float*)d_in[28];
    const float* bd2c2 = (const float*)d_in[29];
    const float* bd2g2 = (const float*)d_in[30]; const float* bd2b2 = (const float*)d_in[31];
    const float* w_dt1 = (const float*)d_in[32];
    const float* w_dt2 = (const float*)d_in[33];

    const int Bn = in_sizes[0] / (3 * 128 * 128);   // 64
    const int Mq = Bn * 32 * 32;                    // latent pixels (65536)

    float* out = (float*)d_out;
    const size_t XP = (size_t)Bn * 3 * 128 * 128;
    const size_t OH = (size_t)Mq * 512;
    float* out_xpred = out;
    float* out_oh    = out + XP;
    float* out_loss  = out + XP + OH;
    float* out_idx   = out + XP + OH + 1;

    // ---- workspace carve ----
    char* wsb = (char*)d_ws; size_t off = 0;
    auto alloc = [&](size_t bytes) -> void* {
        void* p = wsb + off; off = (off + bytes + 511) & ~(size_t)511; return p;
    };
    auto fragsz = [](int O, int I, int k) -> size_t {
        return (size_t)(O / 16) * k * k * (I / 32) * 512 * sizeof(_Float16);
    };
    _Float16* f_ec2   = (_Float16*)alloc(fragsz(32, 32, 3));
    _Float16* f_be1c1 = (_Float16*)alloc(fragsz(32, 32, 3));
    _Float16* f_be1c2 = (_Float16*)alloc(fragsz(32, 32, 3));
    _Float16* f_be2c1 = (_Float16*)alloc(fragsz(32, 32, 3));
    _Float16* f_be2c2 = (_Float16*)alloc(fragsz(32, 32, 3));
    _Float16* f_ec3   = (_Float16*)alloc(fragsz(64, 32, 1));
    _Float16* f_bd1c1 = (_Float16*)alloc(fragsz(32, 64, 3));
    _Float16* f_bd1c2 = (_Float16*)alloc(fragsz(32, 32, 3));
    _Float16* f_sc    = (_Float16*)alloc(fragsz(32, 64, 1));
    _Float16* f_bd2c1 = (_Float16*)alloc(fragsz(32, 32, 3));
    _Float16* f_bd2c2 = (_Float16*)alloc(fragsz(32, 32, 3));
    _Float16* f_dt1   = (_Float16*)alloc(fragsz(32, 32, 3));
    _Float16* f_emb   = (_Float16*)alloc(fragsz(512, 64, 1));
    float*    e2      = (float*)alloc(512 * sizeof(float));
    _Float16* h1      = (_Float16*)alloc((size_t)Bn * 64 * 64 * 32 * sizeof(_Float16));
    _Float16* tIn     = (_Float16*)alloc((size_t)Mq * 32 * sizeof(_Float16));
    _Float16* tA      = (_Float16*)alloc((size_t)Mq * 32 * sizeof(_Float16));
    _Float16* tB      = (_Float16*)alloc((size_t)Mq * 32 * sizeof(_Float16));
    _Float16* tC      = (_Float16*)alloc((size_t)Mq * 32 * sizeof(_Float16));
    _Float16* ze_h    = (_Float16*)alloc((size_t)Mq * 64 * sizeof(_Float16));
    float*    ze_f    = (float*)alloc((size_t)Mq * 64 * sizeof(float));
    _Float16* zq_h    = (_Float16*)alloc((size_t)Mq * 64 * sizeof(_Float16));
    int*      idx_i   = (int*)alloc((size_t)Mq * sizeof(int));
    _Float16* d1      = (_Float16*)alloc((size_t)Bn * 65 * 65 * 32 * sizeof(_Float16));
    float*    acc     = (float*)alloc(2 * sizeof(float));

    vqvae_zero_acc<<<1, 32, 0, stream>>>(acc);

    // ---- weight fragment prep ----
    auto prep = [&](const float* w, _Float16* f, int O, int I, int k, int dec) {
        int total = (O / 16) * k * k * (I / 32) * 512;
        vqvae_prep_wfrag<<<(total + 255) / 256, 256, 0, stream>>>(w, f, O, I, k, dec);
    };
    prep(w_ec2, f_ec2, 32, 32, 3, 0);
    prep(be1c1, f_be1c1, 32, 32, 3, 0);  prep(be1c2, f_be1c2, 32, 32, 3, 0);
    prep(be2c1, f_be2c1, 32, 32, 3, 0);  prep(be2c2, f_be2c2, 32, 32, 3, 0);
    prep(w_ec3, f_ec3, 64, 32, 1, 0);
    prep(bd1c1, f_bd1c1, 32, 64, 3, 0);  prep(bd1c2, f_bd1c2, 32, 32, 3, 0);
    prep(bd1sc, f_sc, 32, 64, 1, 0);
    prep(bd2c1, f_bd2c1, 32, 32, 3, 0);  prep(bd2c2, f_bd2c2, 32, 32, 3, 0);
    prep(w_dt1, f_dt1, 32, 32, 3, 1);
    prep(Emb, f_emb, 512, 64, 1, 0);
    vqvae_prep_e2<<<2, 256, 0, stream>>>(Emb, e2, 512, 64);

    // ---- encoder ----
    vqvae_conv_ec1<<<8192, 256, 0, stream>>>(x, w_ec1, h1, Bn);

    auto grid_of = [&](int Hout, int Wout, int Cout) {
        int M = Bn * Hout * Wout;
        int mtiles = (M + 15) / 16;
        return dim3((mtiles + 7) / 8, Cout / 16);
    };

    // ec2: 64x64x32 -> 32x32x32, stride2 pad1, GELU
    vqvae_conv_wmma<3, 1, 0><<<grid_of(32, 32, 32), 256, 0, stream>>>(
        h1, f_ec2, nullptr, nullptr, nullptr, tIn, nullptr,
        Bn, 64, 64, 32, 32, 32, 2, 1, 2);
    // be1
    vqvae_conv_wmma<3, 1, 0><<<grid_of(32, 32, 32), 256, 0, stream>>>(
        tIn, f_be1c1, be1g1, be1b1, nullptr, tA, nullptr,
        Bn, 32, 32, 32, 32, 32, 1, 1, 1);
    vqvae_conv_wmma<3, 1, 0><<<grid_of(32, 32, 32), 256, 0, stream>>>(
        tA, f_be1c2, be1g2, be1b2, tIn, tB, nullptr,
        Bn, 32, 32, 32, 32, 32, 1, 1, 3);
    // be2
    vqvae_conv_wmma<3, 1, 0><<<grid_of(32, 32, 32), 256, 0, stream>>>(
        tB, f_be2c1, be2g1, be2b1, nullptr, tA, nullptr,
        Bn, 32, 32, 32, 32, 32, 1, 1, 1);
    vqvae_conv_wmma<3, 1, 0><<<grid_of(32, 32, 32), 256, 0, stream>>>(
        tA, f_be2c2, be2g2, be2b2, tB, tC, nullptr,
        Bn, 32, 32, 32, 32, 32, 1, 1, 3);
    // ec3 1x1: 32 -> 64, no act, fp16 + fp32 outputs
    vqvae_conv_wmma<1, 1, 0><<<grid_of(32, 32, 64), 256, 0, stream>>>(
        tC, f_ec3, nullptr, nullptr, nullptr, ze_h, ze_f,
        Bn, 32, 32, 32, 32, 64, 1, 0, 0);

    // ---- vector quantization ----
    {
        int mtiles = Mq / 16;
        vqvae_vq<<<(mtiles + 7) / 8, 256, 0, stream>>>(ze_h, ze_f, Emb, f_emb, e2,
                                                       zq_h, out_oh, out_idx, idx_i,
                                                       acc + 1, Mq);
    }

    // ---- decoder (input z_ss == z_q numerically) ----
    vqvae_conv_wmma<3, 2, 0><<<grid_of(32, 32, 32), 256, 0, stream>>>(
        zq_h, f_bd1c1, bd1g1, bd1b1, nullptr, tA, nullptr,
        Bn, 32, 32, 32, 32, 32, 1, 1, 1);
    vqvae_conv_wmma<1, 2, 0><<<grid_of(32, 32, 32), 256, 0, stream>>>(
        zq_h, f_sc, bd1sg, bd1sb, nullptr, tIn, nullptr,
        Bn, 32, 32, 32, 32, 32, 1, 0, 0);
    vqvae_conv_wmma<3, 1, 0><<<grid_of(32, 32, 32), 256, 0, stream>>>(
        tA, f_bd1c2, bd1g2, bd1b2, tIn, tB, nullptr,
        Bn, 32, 32, 32, 32, 32, 1, 1, 3);
    vqvae_conv_wmma<3, 1, 0><<<grid_of(32, 32, 32), 256, 0, stream>>>(
        tB, f_bd2c1, bd2g1, bd2b1, nullptr, tA, nullptr,
        Bn, 32, 32, 32, 32, 32, 1, 1, 1);
    vqvae_conv_wmma<3, 1, 0><<<grid_of(32, 32, 32), 256, 0, stream>>>(
        tA, f_bd2c2, bd2g2, bd2b2, tB, tC, nullptr,
        Bn, 32, 32, 32, 32, 32, 1, 1, 3);
    // dt1 deconv: 32x32 -> 65x65, dilated-input conv with pad_lo 2, GELU
    vqvae_conv_wmma<3, 1, 1><<<grid_of(65, 65, 32), 256, 0, stream>>>(
        tC, f_dt1, nullptr, nullptr, nullptr, d1, nullptr,
        Bn, 32, 32, 65, 65, 32, 1, 2, 2);

    // dt2 + sigmoid + x_pred + MSE partials
    vqvae_deconv_final<<<4096, 256, 0, stream>>>(d1, w_dt2, x, out_xpred, acc, Bn);

    vqvae_finalize<<<1, 32, 0, stream>>>(acc, out_loss,
                                         (float)((size_t)Bn * 3 * 128 * 128),
                                         (float)((size_t)Mq * 64),
                                         1.f / (float)Bn);
}
